// Block_26895085207779
// MI455X (gfx1250) — compile-verified, matrix operand order checked by reading, not run on hardware
//
#include <hip/hip_runtime.h>
#include <hip/hip_bf16.h>

// ---------------------------------------------------------------------------
// Types
// ---------------------------------------------------------------------------
typedef __attribute__((ext_vector_type(16))) __bf16 bf16x16;
typedef __attribute__((ext_vector_type(8)))  float  f32x8;
typedef __attribute__((ext_vector_type(4)))  unsigned int u32x4;
typedef __attribute__((ext_vector_type(8)))  int    i32x8;
typedef __attribute__((ext_vector_type(4)))  int    i32x4;

union FragBF {            // one WMMA A/B fragment = 16 bf16 per lane = 2 x 16B
    uint4   q[2];
    bf16x16 v;
};

__device__ __forceinline__ unsigned short f2bf(float f) {
    unsigned u = __float_as_uint(f);
    unsigned r = (u + 0x7FFFu + ((u >> 16) & 1u)) >> 16;   // round-nearest-even
    return (unsigned short)r;
}

// ---------------------------------------------------------------------------
// Problem constants (B=4, L=2048, E=1024)
// ---------------------------------------------------------------------------
#define BATCH   4
#define SEQL    2048
#define EDIM    1024
#define DIN     2048          // d_inner
#define DST     16            // d_state
#define DTR     64            // dt_rank
#define HFFN    4096
#define MROWS   (BATCH * SEQL)   // 8192

// ---------------------------------------------------------------------------
// TDM descriptor issue: 2-D tile (BK x rows) of bf16, global -> LDS,
// with hardware row padding reproducing the LDK=40 (80B) LDS stride.
//   tensor_dim0 = k_rem  (elements left in K from this k0; OOB -> zeros)
//   tensor_dim1 = rows_rem (rows left in the matrix;       OOB -> zeros)
//   dim0 stride = K elements; pad 16B after every 64B stored.
// ---------------------------------------------------------------------------
#define BM 128
#define BN 128
#define BK 32
#define LDK 40                 // padded LDS K-stride (bf16 elems), 80B rows

__device__ __forceinline__ void tdm_load_tile(unsigned lds_addr,
                                              const unsigned short* gptr,
                                              unsigned rows_rem,
                                              unsigned k_rem,
                                              unsigned K)
{
    const unsigned long long ga = (unsigned long long)(size_t)gptr;
    u32x4 g0;
    g0[0] = 1u;                                            // count=1, user mode
    g0[1] = lds_addr;                                      // LDS byte address
    g0[2] = (unsigned)(ga & 0xFFFFFFFFull);                // global_addr[31:0]
    g0[3] = (unsigned)((ga >> 32) & 0x01FFFFFFull)         // global_addr[56:32]
          | (2u << 30);                                    // type = 2 (image)

    i32x8 g1;
    // data_size=2B (1<<16) | pad_enable (1<<20) | pad_interval=3 -> 64B (3<<22)
    // | pad_amount=3 -> 4 dwords = 16B (3<<25)
    g1[0] = (int)((1u << 16) | (1u << 20) | (3u << 22) | (3u << 25));
    g1[1] = (int)(k_rem << 16);                            // td0[15:0] @ bits[63:48]
    g1[2] = (int)((k_rem >> 16) | (rows_rem << 16));       // td0[31:16] | td1[15:0]
    g1[3] = (int)((rows_rem >> 16) | ((unsigned)BK << 16));// td1[31:16] | tile_dim0
    g1[4] = (int)(unsigned)BM;                             // tile_dim1=128, tile_dim2=0
    g1[5] = (int)K;                                        // dim0_stride[31:0]
    g1[6] = 0;                                             // dim0_stride[47:32] | st1
    g1[7] = 0;

    i32x4 z4; z4[0] = 0; z4[1] = 0; z4[2] = 0; z4[3] = 0;  // groups 2/3: NULL (2-D)
    i32x8 z8;
    #pragma unroll
    for (int i = 0; i < 8; ++i) z8[i] = 0;
    // amdgpu-toolchain (clang-23) 6-arg form: (g0, g1, g2, g3, extra, cpol)
    __builtin_amdgcn_tensor_load_to_lds(g0, g1, z4, z4, z8, 0);
}

// ---------------------------------------------------------------------------
// Generic bf16 WMMA GEMM:  C[M,N] = A[M,K] * Bw[N,K]^T  (+bias,+relu,+resid)
//   block tile 128x128x32, 256 threads = 8 waves, wave tile 64x32 (4x2 WMMA)
//   TDM double-buffered staging: wave 0 issues DMA for tile i+1 while all
//   waves run WMMA on tile i.
// ---------------------------------------------------------------------------
__global__ void __launch_bounds__(256)
gemm_bf16_wmma(const unsigned short* __restrict__ A,
               const unsigned short* __restrict__ Bw,
               float* __restrict__ C,                 // nullable
               unsigned short* __restrict__ Cbf,      // nullable
               const float* __restrict__ bias,        // nullable, len N
               const float* __restrict__ resid,       // nullable, [M,N]
               int M, int N, int K, int relu)
{
    __shared__ unsigned short As[2][BM * LDK];
    __shared__ unsigned short Bs[2][BN * LDK];

    const int tid  = threadIdx.x;
    const int wave = tid >> 5;
    const int lane = tid & 31;
    const int half = lane >> 4;      // which 16-lane half of the wave
    const int lr   = lane & 15;

    const int wm = wave & 1;         // wave grid 2 (M) x 4 (N)
    const int wn = wave >> 1;
    const int block_m = blockIdx.y * BM;
    const int block_n = blockIdx.x * BN;

    const unsigned rows_a = (unsigned)(M - block_m);
    const unsigned rows_b = (unsigned)(N - block_n);
    const unsigned lds_a0 = (unsigned)(size_t)(void*)&As[0][0];
    const unsigned lds_a1 = (unsigned)(size_t)(void*)&As[1][0];
    const unsigned lds_b0 = (unsigned)(size_t)(void*)&Bs[0][0];
    const unsigned lds_b1 = (unsigned)(size_t)(void*)&Bs[1][0];

    f32x8 acc[4][2];
    #pragma unroll
    for (int i = 0; i < 4; ++i)
        #pragma unroll
        for (int j = 0; j < 2; ++j)
            #pragma unroll
            for (int r = 0; r < 8; ++r) acc[i][j][r] = 0.0f;

    // ---- prologue: DMA tile 0 into buffer 0 ----
    if (wave == 0) {
        tdm_load_tile(lds_a0, A  + (size_t)block_m * K, rows_a, (unsigned)K, (unsigned)K);
        tdm_load_tile(lds_b0, Bw + (size_t)block_n * K, rows_b, (unsigned)K, (unsigned)K);
    }

    int buf = 0;
    for (int k0 = 0; k0 < K; k0 += BK, buf ^= 1) {
        __syncthreads();   // all waves done reading buf^1 (overwritten below)
        if (wave == 0) {
            if (k0 + BK < K) {
                const int kn = k0 + BK;
                const unsigned la = buf ? lds_a0 : lds_a1;
                const unsigned lb = buf ? lds_b0 : lds_b1;
                tdm_load_tile(la, A  + (size_t)block_m * K + kn,
                              rows_a, (unsigned)(K - kn), (unsigned)K);
                tdm_load_tile(lb, Bw + (size_t)block_n * K + kn,
                              rows_b, (unsigned)(K - kn), (unsigned)K);
                __builtin_amdgcn_s_wait_tensorcnt(2);  // pair for tile k0 landed
            } else {
                __builtin_amdgcn_s_wait_tensorcnt(0);  // final tile landed
            }
        }
        __syncthreads();   // tile k0 visible to every wave

        const unsigned short* Ab = &As[buf][0];
        const unsigned short* Bb = &Bs[buf][0];

        // Fragment loads: per ISA 16-bit A layout, lane's 16 bf16 values are
        // k = 8*half + [0..7]  and  k = 8*half + 16 + [0..7] -> 2x ds_load_b128
        FragBF a[4], b[2];
        #pragma unroll
        for (int mt = 0; mt < 4; ++mt) {
            const int base = (wm * 64 + mt * 16 + lr) * LDK + half * 8;
            a[mt].q[0] = *(const uint4*)&Ab[base];
            a[mt].q[1] = *(const uint4*)&Ab[base + 16];
        }
        #pragma unroll
        for (int nt = 0; nt < 2; ++nt) {
            const int base = (wn * 32 + nt * 16 + lr) * LDK + half * 8;
            b[nt].q[0] = *(const uint4*)&Bb[base];
            b[nt].q[1] = *(const uint4*)&Bb[base + 16];
        }

        #pragma unroll
        for (int mt = 0; mt < 4; ++mt)
            #pragma unroll
            for (int nt = 0; nt < 2; ++nt)
                acc[mt][nt] = __builtin_amdgcn_wmma_f32_16x16x32_bf16(
                    false, a[mt].v, false, b[nt].v,
                    (short)0, acc[mt][nt], false, false);
    }

    // Epilogue: C/D layout — VGPR r: lanes0-15 -> M=r, lanes16-31 -> M=r+8; N=lr
    #pragma unroll
    for (int mt = 0; mt < 4; ++mt) {
        #pragma unroll
        for (int nt = 0; nt < 2; ++nt) {
            const int col = block_n + wn * 32 + nt * 16 + lr;
            if (col >= N) continue;
            const float bv = bias ? bias[col] : 0.0f;
            #pragma unroll
            for (int r = 0; r < 8; ++r) {
                const int row = block_m + wm * 64 + mt * 16 + half * 8 + r;
                const size_t idx = (size_t)row * N + col;
                float v = acc[mt][nt][r] + bv;
                if (relu) v = fmaxf(v, 0.0f);
                if (resid) v += resid[idx];
                if (C)   C[idx]   = v;
                if (Cbf) Cbf[idx] = f2bf(v);
            }
        }
    }
}

// ---------------------------------------------------------------------------
// LayerNorm (one block per row of E=1024) -> bf16
// ---------------------------------------------------------------------------
__global__ void __launch_bounds__(256)
layernorm_bf16_kernel(const float* __restrict__ x,
                      const float* __restrict__ g, const float* __restrict__ b,
                      unsigned short* __restrict__ out, int E)
{
    const int row = blockIdx.x;
    const float* xr = x + (size_t)row * E;
    float s = 0.f, s2 = 0.f;
    for (int i = threadIdx.x; i < E; i += 256) { float v = xr[i]; s += v; s2 += v * v; }
    #pragma unroll
    for (int m = 16; m; m >>= 1) { s += __shfl_xor(s, m, 32); s2 += __shfl_xor(s2, m, 32); }
    __shared__ float rs[8], rs2[8];
    const int wave = threadIdx.x >> 5, lane = threadIdx.x & 31;
    if (lane == 0) { rs[wave] = s; rs2[wave] = s2; }
    __syncthreads();
    if (wave == 0) {
        s  = (lane < 8) ? rs[lane]  : 0.f;
        s2 = (lane < 8) ? rs2[lane] : 0.f;
        #pragma unroll
        for (int m = 4; m; m >>= 1) { s += __shfl_xor(s, m, 32); s2 += __shfl_xor(s2, m, 32); }
        if (lane == 0) { rs[0] = s; rs2[0] = s2; }
    }
    __syncthreads();
    const float mean = rs[0] / E;
    const float var  = rs2[0] / E - mean * mean;
    const float rstd = rsqrtf(var + 1e-5f);
    for (int i = threadIdx.x; i < E; i += 256)
        out[(size_t)row * E + i] = f2bf((xr[i] - mean) * rstd * g[i] + b[i]);
}

// ---------------------------------------------------------------------------
// Elementwise helpers
// ---------------------------------------------------------------------------
__global__ void f32_to_bf16_kernel(const float* __restrict__ in,
                                   unsigned short* __restrict__ out, size_t n)
{
    size_t i = (size_t)blockIdx.x * blockDim.x + threadIdx.x;
    const size_t stride = (size_t)gridDim.x * blockDim.x;
    for (; i < n; i += stride) out[i] = f2bf(in[i]);
}

// dt slice of dbl[8192,96] (cols 0..63) -> bf16 [8192,64]
__global__ void dt_slice_bf16_kernel(const float* __restrict__ dbl,
                                     unsigned short* __restrict__ out)
{
    const size_t idx = (size_t)blockIdx.x * 256 + threadIdx.x;
    if (idx >= (size_t)MROWS * DTR) return;
    const size_t r = idx / DTR, c = idx % DTR;
    out[idx] = f2bf(dbl[r * 96 + c]);
}

// causal depthwise conv (k=4) + bias + SiLU; input = xz[:, :DIN]
__global__ void conv_silu_kernel(const float* __restrict__ xz,
                                 const float* __restrict__ wconv,
                                 const float* __restrict__ bconv,
                                 float* __restrict__ xc,
                                 unsigned short* __restrict__ xc_bf)
{
    const size_t idx = (size_t)blockIdx.x * 256 + threadIdx.x;
    if (idx >= (size_t)MROWS * DIN) return;
    const int d = idx & (DIN - 1);
    const size_t bl = idx >> 11;            // b*L + l
    const int l = (int)(bl & (SEQL - 1));
    float acc = bconv[d];
    #pragma unroll
    for (int k = 0; k < 4; ++k) {
        const int ll = l - 3 + k;
        if (ll >= 0) acc += wconv[k * DIN + d] * xz[(bl - 3 + k) * (2 * DIN) + d];
    }
    const float sv = acc / (1.0f + expf(-acc));
    xc[idx] = sv;
    xc_bf[idx] = f2bf(sv);
}

// delta = softplus(delta_raw + b_dt[d])   (in place)
__global__ void softplus_kernel(float* __restrict__ delta,
                                const float* __restrict__ b_dt)
{
    const size_t idx = (size_t)blockIdx.x * 256 + threadIdx.x;
    if (idx >= (size_t)MROWS * DIN) return;
    const int d = idx & (DIN - 1);
    const float v = delta[idx] + b_dt[d];
    delta[idx] = (v > 20.0f) ? v : log1pf(expf(v));
}

// SSM scan: thread per (b, d, s); 16-lane shuffle reduction over d_state.
__global__ void __launch_bounds__(256)
ssm_scan_kernel(const float* __restrict__ delta,   // [M, DIN]
                const float* __restrict__ xc,      // [M, DIN]
                const float* __restrict__ dbl,     // [M, 96] (B @64, C @80)
                const float* __restrict__ A_log,   // [DIN, 16]
                float* __restrict__ ys)            // [M, DIN]
{
    const int t = blockIdx.x * 256 + threadIdx.x;
    const int s = t & 15;
    const int d = (t >> 4) & (DIN - 1);
    const int b = t >> 15;
    const float Aneg = -expf(A_log[d * DST + s]) * 1.44269504088896340736f; // *log2e
    float h = 0.f;
    for (int l = 0; l < SEQL; ++l) {
        const size_t bl = (size_t)b * SEQL + l;
        const float dt = delta[bl * DIN + d];
        const float xv = xc[bl * DIN + d];
        const float Bv = dbl[bl * 96 + 64 + s];
        const float Cv = dbl[bl * 96 + 80 + s];
        h = h * exp2f(dt * Aneg) + dt * xv * Bv;
        float p = h * Cv;
        p += __shfl_xor(p, 1, 16);
        p += __shfl_xor(p, 2, 16);
        p += __shfl_xor(p, 4, 16);
        p += __shfl_xor(p, 8, 16);
        if (s == 0) ys[bl * DIN + d] = p;
    }
}

// y = (ys + D*xc) * silu(z) -> bf16 (A operand of out-proj GEMM)
__global__ void y_gate_kernel(const float* __restrict__ ys,
                              const float* __restrict__ xc,
                              const float* __restrict__ xz,
                              const float* __restrict__ Dp,
                              unsigned short* __restrict__ y_bf)
{
    const size_t idx = (size_t)blockIdx.x * 256 + threadIdx.x;
    if (idx >= (size_t)MROWS * DIN) return;
    const int d = idx & (DIN - 1);
    const size_t bl = idx >> 11;
    const float z = xz[bl * (2 * DIN) + DIN + d];
    const float v = (ys[idx] + Dp[d] * xc[idx]) * (z / (1.0f + expf(-z)));
    y_bf[idx] = f2bf(v);
}

// ---------------------------------------------------------------------------
// Host orchestration
// ---------------------------------------------------------------------------
extern "C" void kernel_launch(void* const* d_in, const int* in_sizes, int n_in,
                              void* d_out, int out_size, void* d_ws, size_t ws_size,
                              hipStream_t stream)
{
    const float* x      = (const float*)d_in[0];
    const float* w_in   = (const float*)d_in[1];
    const float* w_conv = (const float*)d_in[2];
    const float* b_conv = (const float*)d_in[3];
    const float* w_xprj = (const float*)d_in[4];
    const float* w_dt   = (const float*)d_in[5];
    const float* b_dt   = (const float*)d_in[6];
    const float* A_log  = (const float*)d_in[7];
    const float* Dp     = (const float*)d_in[8];
    const float* w_out  = (const float*)d_in[9];
    const float* ln1_g  = (const float*)d_in[10];
    const float* ln1_b  = (const float*)d_in[11];
    const float* ln2_g  = (const float*)d_in[12];
    const float* ln2_b  = (const float*)d_in[13];
    const float* w_ffn1 = (const float*)d_in[14];
    const float* b_ffn1 = (const float*)d_in[15];
    const float* w_ffn2 = (const float*)d_in[16];
    const float* b_ffn2 = (const float*)d_in[17];
    float* out = (float*)d_out;

    char* ws = (char*)d_ws;
    size_t off = 0;
    auto alloc = [&](size_t bytes) -> void* {
        void* p = ws + off;
        off += (bytes + 255) & ~(size_t)255;
        return p;
    };

    const size_t M = MROWS;
    unsigned short* xln_bf   = (unsigned short*)alloc(M * EDIM * 2);
    unsigned short* w_in_bf  = (unsigned short*)alloc((size_t)2 * DIN * EDIM * 2);
    unsigned short* w_xp_bf  = (unsigned short*)alloc((size_t)96 * DIN * 2);
    unsigned short* w_dt_bf  = (unsigned short*)alloc((size_t)DIN * DTR * 2);
    unsigned short* w_out_bf = (unsigned short*)alloc((size_t)EDIM * DIN * 2);
    unsigned short* w_f1_bf  = (unsigned short*)alloc((size_t)HFFN * EDIM * 2);
    unsigned short* w_f2_bf  = (unsigned short*)alloc((size_t)EDIM * HFFN * 2);
    float*          xz       = (float*)alloc(M * 2 * DIN * 4);       // reused as h1_bf
    float*          xc       = (float*)alloc(M * DIN * 4);
    unsigned short* xc_bf    = (unsigned short*)alloc(M * DIN * 2);  // reused as y_bf
    float*          dbl      = (float*)alloc(M * 96 * 4);
    unsigned short* dt_bf    = (unsigned short*)alloc(M * DTR * 2);
    float*          delta    = (float*)alloc(M * DIN * 4);           // reused as ln2_bf
    float*          ysb      = (float*)alloc(M * DIN * 4);
    float*          x2       = (float*)alloc(M * EDIM * 4);
    // aliases over dead buffers:
    unsigned short* y_bf   = xc_bf;                 // alive after G4 (x-proj) done
    unsigned short* h1_bf  = (unsigned short*)xz;   // alive after y_gate consumed z
    unsigned short* ln2_bf = (unsigned short*)delta;// alive after scan consumed delta
    (void)ws_size; (void)in_sizes; (void)n_in; (void)out_size;

    auto cvt = [&](const float* src, unsigned short* dst, size_t n) {
        const int blocks = (int)(((n + 255) / 256 > 4096) ? 4096 : (n + 255) / 256);
        f32_to_bf16_kernel<<<blocks, 256, 0, stream>>>(src, dst, n);
    };

    // 1. weight conversions
    cvt(w_in,   w_in_bf,  (size_t)2 * DIN * EDIM);
    cvt(w_xprj, w_xp_bf,  (size_t)96 * DIN);
    cvt(w_dt,   w_dt_bf,  (size_t)DIN * DTR);
    cvt(w_out,  w_out_bf, (size_t)EDIM * DIN);
    cvt(w_ffn1, w_f1_bf,  (size_t)HFFN * EDIM);
    cvt(w_ffn2, w_f2_bf,  (size_t)EDIM * HFFN);

    const dim3 blk(256);
    auto gemm = [&](const unsigned short* A, const unsigned short* Bw,
                    float* C, unsigned short* Cbf, const float* bias,
                    const float* resid, int m, int n, int k, int relu) {
        dim3 grid((n + BN - 1) / BN, (m + BM - 1) / BM);
        gemm_bf16_wmma<<<grid, blk, 0, stream>>>(A, Bw, C, Cbf, bias, resid, m, n, k, relu);
    };

    // 2. LN1 -> bf16
    layernorm_bf16_kernel<<<(int)M, blk, 0, stream>>>(x, ln1_g, ln1_b, xln_bf, EDIM);

    // 3. in-proj: xz = ln1(x) @ w_in^T        [8192 x 4096]
    gemm(xln_bf, w_in_bf, xz, nullptr, nullptr, nullptr, (int)M, 2 * DIN, EDIM, 0);

    // 4. causal conv + SiLU
    conv_silu_kernel<<<(int)(M * DIN / 256), blk, 0, stream>>>(xz, w_conv, b_conv, xc, xc_bf);

    // 5. x-proj: dbl = xc @ w_xproj^T          [8192 x 96]
    gemm(xc_bf, w_xp_bf, dbl, nullptr, nullptr, nullptr, (int)M, 96, DIN, 0);

    // 6. dt slice -> bf16
    dt_slice_bf16_kernel<<<(int)(M * DTR / 256), blk, 0, stream>>>(dbl, dt_bf);

    // 7. dt-proj: delta_raw = dt @ w_dt^T      [8192 x 2048]
    gemm(dt_bf, w_dt_bf, delta, nullptr, nullptr, nullptr, (int)M, DIN, DTR, 0);

    // 8. softplus(+b_dt) in place
    softplus_kernel<<<(int)(M * DIN / 256), blk, 0, stream>>>(delta, b_dt);

    // 9. selective scan (parallel over b,d,s; shfl-reduce over d_state)
    ssm_scan_kernel<<<(int)(BATCH * DIN * DST / 256), blk, 0, stream>>>(
        delta, xc, dbl, A_log, ysb);

    // 10. y = (ys + D*xc) * silu(z) -> bf16
    y_gate_kernel<<<(int)(M * DIN / 256), blk, 0, stream>>>(ysb, xc, xz, Dp, y_bf);

    // 11. out-proj + residual(x): x2 = x + y @ w_out^T   [8192 x 1024]
    gemm(y_bf, w_out_bf, x2, nullptr, nullptr, x, (int)M, EDIM, DIN, 0);

    // 12. LN2 -> bf16
    layernorm_bf16_kernel<<<(int)M, blk, 0, stream>>>(x2, ln2_g, ln2_b, ln2_bf, EDIM);

    // 13. FFN1: h1 = relu(ln2 @ w_ffn1^T + b1) -> bf16 only  [8192 x 4096]
    gemm(ln2_bf, w_f1_bf, nullptr, h1_bf, b_ffn1, nullptr, (int)M, HFFN, EDIM, 1);

    // 14. FFN2 + bias + residual(x2) -> output f32          [8192 x 1024]
    gemm(h1_bf, w_f2_bf, out, nullptr, b_ffn2, x2, (int)M, EDIM, HFFN, 0);
}
